// FeatureFusionModule_73650099191962
// MI455X (gfx1250) — compile-verified, matrix operand order checked by reading, not run on hardware
//
#include <hip/hip_runtime.h>
#include <math.h>

#define NB   16
#define CCH  64
#define HH   128
#define WW   128
#define HWP  (HH*WW)

typedef float    v2f  __attribute__((ext_vector_type(2)));
typedef float    v8f  __attribute__((ext_vector_type(8)));
typedef _Float16 v8h  __attribute__((ext_vector_type(8)));
typedef _Float16 v16h __attribute__((ext_vector_type(16)));

// ---------------------------------------------------------------- row sums
__global__ void rowsum_k(const float* __restrict__ img, const float* __restrict__ dep,
                         float* __restrict__ sums) {
  int n = blockIdx.x, r = blockIdx.y;
  const float* src = (r < CCH) ? img + ((size_t)(n*CCH + r))*HWP
                               : dep + ((size_t)(n*CCH + (r-CCH)))*HWP;
  float s = 0.f;
  for (int i = threadIdx.x; i < HWP; i += 256) s += src[i];
  __shared__ float red[256];
  red[threadIdx.x] = s; __syncthreads();
  for (int off = 128; off > 0; off >>= 1) {
    if (threadIdx.x < off) red[threadIdx.x] += red[threadIdx.x + off];
    __syncthreads();
  }
  if (threadIdx.x == 0) sums[n*128 + r] = red[0];
}

// ------------------------------------------------- Gram G = X * Y^T (f32 WMMA)
__global__ void gram_k(const float* __restrict__ img, const float* __restrict__ dep,
                       float* __restrict__ G) {
  __shared__ float Xs[64*64];
  __shared__ float Ys[64*64];
  int n = blockIdx.x, tid = threadIdx.x;
  int lane = tid & 31, wave = tid >> 5;
  int half = lane >> 4, m = lane & 15;
  int a0 = (wave >> 2) * 16, b0 = (wave & 3) * 16;
  const float* X = img + (size_t)n*CCH*HWP;
  const float* Y = dep + (size_t)n*CCH*HWP;
  v8f acc = {};
  for (int kc = 0; kc < HWP; kc += 64) {
    for (int j = 0; j < 8; j++) {
      int i = tid + j*512; int row = i >> 6, col = i & 63;
      Xs[i] = X[(size_t)row*HWP + kc + col];
      Ys[i] = Y[(size_t)row*HWP + kc + col];
    }
    if (kc + 64 < HWP) {   // pull next chunk toward the caches
      int row = tid & 63;
      __builtin_prefetch(&X[(size_t)row*HWP + kc + 64], 0, 0);
      __builtin_prefetch(&Y[(size_t)row*HWP + kc + 64], 0, 0);
    }
    __syncthreads();
    for (int kk = 0; kk < 16; kk++) {
      int kb = kk*4 + half*2;
      v2f a, b;
      a.x = Xs[(a0+m)*64 + kb]; a.y = Xs[(a0+m)*64 + kb + 1];
      b.x = Ys[(b0+m)*64 + kb]; b.y = Ys[(b0+m)*64 + kb + 1];
      acc = __builtin_amdgcn_wmma_f32_16x16x4_f32(false, a, false, b, (short)0, acc, false, false);
    }
    __syncthreads();
  }
  for (int g = 0; g < 8; g++) {
    int row = a0 + g + half*8, col = b0 + m;
    G[(size_t)n*4096 + row*64 + col] = acc[g];
  }
}

// ----------------- tiny 64x64 algebra: logits, softmax, M = gamma*attn*Wv
__global__ void attn_k(const float* __restrict__ G, const float* __restrict__ sums,
                       const float* __restrict__ Wq, const float* __restrict__ bq,
                       const float* __restrict__ Wk, const float* __restrict__ bk,
                       const float* __restrict__ Wv, const float* __restrict__ bv,
                       const float* __restrict__ gamma,
                       float* __restrict__ gM1, float* __restrict__ gM2,
                       float* __restrict__ gc1, float* __restrict__ gc2) {
  __shared__ float sT1[4096], sT2[4096];
  __shared__ float s4[256];
  int n = blockIdx.x, t = threadIdx.x;
  const float* Gn = G + (size_t)n*4096;
  const float* sx = sums + n*128;
  const float* sy = sx + 64;
  if (t < 64) {
    float a=0,b=0,c=0,d=0;
    for (int i = 0; i < 64; i++) {
      float wq = Wq[t*64+i], wk = Wk[t*64+i];
      a += wq*sx[i]; b += wq*sy[i]; c += wk*sx[i]; d += wk*sy[i];
    }
    s4[t]=a; s4[64+t]=b; s4[128+t]=c; s4[192+t]=d;
  }
  for (int e = t; e < 4096; e += 256) {          // T1 = Wq*G ; T2 = Wq*G^T
    int cc = e >> 6, bb = e & 63;
    float t1=0, t2=0;
    for (int a = 0; a < 64; a++) {
      float w = Wq[cc*64+a];
      t1 += w*Gn[a*64+bb]; t2 += w*Gn[bb*64+a];
    }
    sT1[e]=t1; sT2[e]=t2;
  }
  __syncthreads();
  float l1[16], l2[16];
  for (int i = 0; i < 16; i++) {                  // logits = T*Wk^T + biases
    int e = t + 256*i; int cc = e >> 6, dd = e & 63;
    float v1=0, v2=0;
    for (int b = 0; b < 64; b++) {
      float wk = Wk[dd*64+b];
      v1 += sT1[cc*64+b]*wk; v2 += sT2[cc*64+b]*wk;
    }
    float bb4 = (float)HWP*bq[cc]*bk[dd];
    v1 += bk[dd]*s4[cc]      + bq[cc]*s4[192+dd] + bb4;
    v2 += bk[dd]*s4[64+cc]   + bq[cc]*s4[128+dd] + bb4;
    l1[i]=v1; l2[i]=v2;
  }
  __syncthreads();
  for (int i = 0; i < 16; i++) { int e = t + 256*i; sT1[e]=l1[i]; sT2[e]=l2[i]; }
  __syncthreads();
  if (t < 128) {                                  // row softmax
    float* row = (t < 64) ? &sT1[t*64] : &sT2[(t-64)*64];
    float mx = row[0];
    for (int i = 1; i < 64; i++) mx = fmaxf(mx, row[i]);
    float s = 0.f;
    for (int i = 0; i < 64; i++) { float ev = __expf(row[i]-mx); row[i]=ev; s+=ev; }
    float inv = 1.f/s;
    for (int i = 0; i < 64; i++) row[i] *= inv;
  }
  __syncthreads();
  float gm = gamma[0];
  for (int e = t; e < 4096; e += 256) {           // M = gamma * attn * Wv
    int cc = e >> 6, bb = e & 63;
    float m1=0, m2=0;
    for (int d = 0; d < 64; d++) {
      float wv = Wv[d*64+bb];
      m1 += sT1[cc*64+d]*wv; m2 += sT2[cc*64+d]*wv;
    }
    gM1[(size_t)n*4096+e] = gm*m1; gM2[(size_t)n*4096+e] = gm*m2;
  }
  if (t < 64) {
    float c1=0, c2=0;
    for (int d = 0; d < 64; d++) { c1 += sT1[t*64+d]*bv[d]; c2 += sT2[t*64+d]*bv[d]; }
    gc1[n*64+t] = gm*c1; gc2[n*64+t] = gm*c2;
  }
}

// ---- apply: att = gM*src + base + gc (f32 WMMA); output NHWC f16 conv input
__global__ void apply_k(const float* __restrict__ img, const float* __restrict__ dep,
                        const float* __restrict__ gM1, const float* __restrict__ gM2,
                        const float* __restrict__ gc1, const float* __restrict__ gc2,
                        _Float16* __restrict__ fh) {
  int n = blockIdx.x;
  int p0 = blockIdx.y * 16;
  int tid = threadIdx.x;
  int lane = tid & 31, wave = tid >> 5;
  int half = lane >> 4, m = lane & 15;
  int dir = wave >> 2;                 // 0: img_att (src=dep), 1: dep_att (src=img)
  int c0 = (wave & 3) * 16;
  const float* M    = (dir == 0 ? gM1 : gM2) + (size_t)n*4096;
  const float* gc   = (dir == 0 ? gc1 : gc2) + n*64;
  const float* src  = (dir == 0 ? dep : img) + (size_t)n*CCH*HWP;
  const float* base = (dir == 0 ? img : dep) + (size_t)n*CCH*HWP;
  v8f acc = {};
  for (int k0 = 0; k0 < 64; k0 += 4) {
    int kb = k0 + half*2;
    v2f a, b;
    a.x = M[(c0+m)*64 + kb]; a.y = M[(c0+m)*64 + kb + 1];
    b.x = src[(size_t)kb*HWP + p0 + m];
    b.y = src[(size_t)(kb+1)*HWP + p0 + m];
    acc = __builtin_amdgcn_wmma_f32_16x16x4_f32(false, a, false, b, (short)0, acc, false, false);
  }
  int p = p0 + m;
  v8h hv;
  for (int g = 0; g < 8; g++) {
    int c = c0 + half*8 + g;
    float v = acc[g] + base[(size_t)c*HWP + p] + gc[c];
    hv[g] = (_Float16)v;
  }
  // NHWC: fh[n][pixel][128]; 8 consecutive channels -> one 16B store per lane
  *(v8h*)&fh[((size_t)(n*HWP + p))*128 + dir*64 + c0 + half*8] = hv;
}

// ----- pack conv weights directly into WMMA B-fragment layout:
// Whf[(tap*4+icc)*4 + otile][lane][16 halves] -> one contiguous v16h per lane
__global__ void wconv_k(const float* __restrict__ Wf, _Float16* __restrict__ Whf) {
  int i = blockIdx.x*256 + threadIdx.x;
  if (i >= 36*4*32*16) return;
  int e     = i & 15;
  int lane  = (i >> 4) & 31;
  int ot    = (i >> 9) & 3;
  int chunk = i >> 11;            // 0..35
  int tap = chunk >> 2, icc = chunk & 3;
  int half = lane >> 4;
  int k  = (e & 7) + ((e >> 3) << 4) + (half << 3);
  int ic = icc*32 + k;
  int o  = ot*16 + (lane & 15);
  Whf[i] = (_Float16)Wf[(size_t)o*1152 + ic*9 + tap];
}

// --------- implicit-GEMM 3x3 conv 128->64 + bias + BN + ReLU (f16 WMMA)
// NHWC input; LDS tile [3 rows][34 cols][128 ch] so fragments are contiguous.
__global__ void conv_k(const _Float16* __restrict__ fh, const _Float16* __restrict__ Whf,
                       const float* __restrict__ bf,  const float* __restrict__ bng,
                       const float* __restrict__ bnb, const float* __restrict__ bnm,
                       const float* __restrict__ bnv, float* __restrict__ out) {
  __shared__ __align__(16) _Float16 Ls[3*34*128];   // 26112 B
  int n = blockIdx.x, y = blockIdx.y, x0w = blockIdx.z * 32;
  int tid = threadIdx.x;
  const _Float16* fin = fh + (size_t)n*HWP*128;
  // stage: 1632 v8h chunks, contiguous 256B per pixel in global (NHWC)
  for (int s8 = tid; s8 < 3*34*16; s8 += 256) {
    int pix = s8 >> 4;                 // 0..101
    int ic0 = (s8 & 15) * 8;
    int dyy = pix / 34, xx = pix % 34;
    int gy = y + dyy - 1, gx = x0w - 1 + xx;
    v8h v = {};
    if (gy >= 0 && gy < HH && gx >= 0 && gx < WW)
      v = *(const v8h*)&fin[((size_t)(gy*WW + gx))*128 + ic0];
    *(v8h*)&Ls[s8*8] = v;
  }
  __syncthreads();
  int lane = tid & 31, wave = tid >> 5;
  int half = lane >> 4, m = lane & 15;
  int x0   = x0w + (wave & 1) * 16;
  int o0   = (wave >> 1) * 16;
  int xoff = (wave & 1) * 16;
  int ot   = wave >> 1;
  v8f acc = {};
  for (int tap = 0; tap < 9; tap++) {
    int dy = tap / 3, dx = tap % 3;
    int col = (dy*34 + xoff + m + dx) * 128;
    #pragma unroll
    for (int icc = 0; icc < 4; icc++) {
      int base = col + icc*32 + half*8;
      v8h alo = *(const v8h*)&Ls[base];         // k 0..7   of this half
      v8h ahi = *(const v8h*)&Ls[base + 16];    // k 16..23 of this half
      v16h a = __builtin_shufflevector(alo, ahi, 0,1,2,3,4,5,6,7,8,9,10,11,12,13,14,15);
      v16h b = *(const v16h*)&Whf[(size_t)(((tap*4 + icc)*4 + ot)*32 + lane)*16];
      acc = __builtin_amdgcn_wmma_f32_16x16x32_f16(false, a, false, b, (short)0, acc, false, false);
    }
  }
  for (int g = 0; g < 8; g++) {
    int px = x0 + g + half*8, oc = o0 + m;
    float inv = bng[oc] * rsqrtf(bnv[oc] + 1e-5f);
    float v = acc[g] + bf[oc];
    v = v * inv + (bnb[oc] - bnm[oc]*inv);
    out[((size_t)(n*CCH + oc))*HWP + y*WW + px] = fmaxf(v, 0.f);
  }
}

// ------------------------------- CBAM: channel pooling
__global__ void pool_k(const float* __restrict__ out, float* __restrict__ pooled) {
  int n = blockIdx.x, c = blockIdx.y;
  const float* src = out + ((size_t)(n*CCH + c))*HWP;
  float s = 0.f, mx = -3.4e38f;
  for (int i = threadIdx.x; i < HWP; i += 256) { float v = src[i]; s += v; mx = fmaxf(mx, v); }
  __shared__ float rs[256], rm[256];
  rs[threadIdx.x] = s; rm[threadIdx.x] = mx; __syncthreads();
  for (int off = 128; off > 0; off >>= 1) {
    if (threadIdx.x < off) {
      rs[threadIdx.x] += rs[threadIdx.x + off];
      rm[threadIdx.x]  = fmaxf(rm[threadIdx.x], rm[threadIdx.x + off]);
    }
    __syncthreads();
  }
  if (threadIdx.x == 0) {
    pooled[n*64 + c]        = rs[0] / (float)HWP;
    pooled[1024 + n*64 + c] = rm[0];
  }
}

// ------------------------------- CBAM: channel MLP + sigmoid
__global__ void mlp_k(const float* __restrict__ pooled,
                      const float* __restrict__ fc1w, const float* __restrict__ fc1b,
                      const float* __restrict__ fc2w, const float* __restrict__ fc2b,
                      float* __restrict__ chs) {
  __shared__ float h[16];
  int n = blockIdx.x, t = threadIdx.x;
  const float* avg = pooled + n*64;
  const float* mxp = pooled + 1024 + n*64;
  if (t < 16) {
    int j = t & 7; const float* src = (t < 8) ? avg : mxp;
    float s = fc1b[j];
    for (int a = 0; a < 64; a++) s += fc1w[j*64+a]*src[a];
    h[t] = fmaxf(s, 0.f);
  }
  __syncthreads();
  float v = 2.f*fc2b[t];
  for (int j = 0; j < 8; j++) v += fc2w[t*8+j]*(h[j] + h[8+j]);
  chs[n*64+t] = 1.f/(1.f + __expf(-v));
}

// ---------- channel scale in place + spatial mean/max over channels
__global__ void sprep_k(float* __restrict__ out, const float* __restrict__ chs,
                        float* __restrict__ sp) {
  int n = blockIdx.x;
  int p = blockIdx.y*256 + threadIdx.x;
  float s = 0.f, mx = -3.4e38f;
  for (int c = 0; c < 64; c++) {
    float v = out[((size_t)(n*64+c))*HWP + p] * chs[n*64+c];
    out[((size_t)(n*64+c))*HWP + p] = v;
    s += v; mx = fmaxf(mx, v);
  }
  sp[((size_t)(n*2))*HWP   + p] = s/64.f;
  sp[((size_t)(n*2+1))*HWP + p] = mx;
}

// ---------- 7x7 spatial conv (2ch -> 1) + sigmoid gate, multiply in place
__global__ void gate_k(float* __restrict__ out, const float* __restrict__ sp,
                       const float* __restrict__ saw, const float* __restrict__ sab) {
  int n = blockIdx.x;
  int p = blockIdx.y*256 + threadIdx.x;
  int y = p >> 7, x = p & 127;
  float acc = sab[0];
  for (int ch = 0; ch < 2; ch++)
    for (int dy = -3; dy <= 3; dy++) {
      int yy = y + dy; if (yy < 0 || yy >= HH) continue;
      for (int dx = -3; dx <= 3; dx++) {
        int xx = x + dx; if (xx < 0 || xx >= WW) continue;
        acc += saw[ch*49 + (dy+3)*7 + (dx+3)] * sp[((size_t)(n*2+ch))*HWP + yy*WW + xx];
      }
    }
  float g = 1.f/(1.f + __expf(-acc));
  for (int c = 0; c < 64; c++) out[((size_t)(n*64+c))*HWP + p] *= g;
}

extern "C" void kernel_launch(void* const* d_in, const int* in_sizes, int n_in,
                              void* d_out, int out_size, void* d_ws, size_t ws_size,
                              hipStream_t stream) {
  const float* img  = (const float*)d_in[0];
  const float* dep  = (const float*)d_in[1];
  const float* Wq   = (const float*)d_in[2];  const float* bq  = (const float*)d_in[3];
  const float* Wk   = (const float*)d_in[4];  const float* bk  = (const float*)d_in[5];
  const float* Wv   = (const float*)d_in[6];  const float* bv  = (const float*)d_in[7];
  const float* gam  = (const float*)d_in[8];
  const float* Wf   = (const float*)d_in[9];  const float* bf  = (const float*)d_in[10];
  const float* bng  = (const float*)d_in[11]; const float* bnb = (const float*)d_in[12];
  const float* bnm  = (const float*)d_in[13]; const float* bnv = (const float*)d_in[14];
  const float* fc1w = (const float*)d_in[15]; const float* fc1b = (const float*)d_in[16];
  const float* fc2w = (const float*)d_in[17]; const float* fc2b = (const float*)d_in[18];
  const float* saw  = (const float*)d_in[19]; const float* sab  = (const float*)d_in[20];
  float* out = (float*)d_out;

  char* ws = (char*)d_ws;
  float*    G      = (float*)   (ws + 0);         // 16*4096 f32      = 256 KB
  float*    sums   = (float*)   (ws + 262144);    // 16*128 f32       = 8 KB
  float*    gM1    = (float*)   (ws + 270336);    // 16*4096 f32      = 256 KB
  float*    gM2    = (float*)   (ws + 532480);    // 256 KB
  float*    gc1    = (float*)   (ws + 794624);    // 4 KB
  float*    gc2    = (float*)   (ws + 798720);    // 4 KB
  _Float16* Whf    = (_Float16*)(ws + 802816);    // 36*4*32*16 f16   = 144 KB
  float*    sp     = (float*)   (ws + 950272);    // 16*2*HW f32      = 2 MB
  float*    pooled = (float*)   (ws + 3047424);   // 2048 f32
  float*    chs    = (float*)   (ws + 3055616);   // 1024 f32
  _Float16* fh     = (_Float16*)(ws + 3059712);   // 16*HW*128 f16 (NHWC) = 64 MB

  rowsum_k<<<dim3(NB, 128), 256, 0, stream>>>(img, dep, sums);
  gram_k  <<<dim3(NB), 512, 0, stream>>>(img, dep, G);
  attn_k  <<<dim3(NB), 256, 0, stream>>>(G, sums, Wq, bq, Wk, bk, Wv, bv, gam,
                                         gM1, gM2, gc1, gc2);
  apply_k <<<dim3(NB, HWP/16), 256, 0, stream>>>(img, dep, gM1, gM2, gc1, gc2, fh);
  wconv_k <<<dim3((36*4*32*16 + 255)/256), 256, 0, stream>>>(Wf, Whf);
  conv_k  <<<dim3(NB, HH, WW/32), 256, 0, stream>>>(fh, Whf, bf, bng, bnb, bnm, bnv, out);
  pool_k  <<<dim3(NB, CCH), 256, 0, stream>>>(out, pooled);
  mlp_k   <<<dim3(NB), 64, 0, stream>>>(pooled, fc1w, fc1b, fc2w, fc2b, chs);
  sprep_k <<<dim3(NB, HWP/256), 256, 0, stream>>>(out, chs, sp);
  gate_k  <<<dim3(NB, HWP/256), 256, 0, stream>>>(out, sp, saw, sab);
}
